// GNOBlock_11227044512422
// MI455X (gfx1250) — matrix operand order repaired and checked by reading, not compiled, not run on hardware
//
#include <hip/hip_runtime.h>
#include <hip/hip_bf16.h>
#include <math.h>

// ---------------- problem constants (fixed by the reference) ----------------
#define NPTS   16384      // source points (y)
#define MPTS   16384      // query points (x)
#define DIMS   3
#define CCH    64         // channels
#define KNBR   16         // max neighbors (== WMMA M dimension!)
#define RADIUS 0.07f
#define NFREQ  32
#define EMBED  192        // D * NFREQ * 2
#define IN0    384        // 2*EMBED
#define H1     128
#define H2     256
#define H3     128
// ln(10000)/32
#define FREQ_SCALE 0.28782313662428553f

typedef __attribute__((ext_vector_type(16))) _Float16 v16h;
typedef __attribute__((ext_vector_type(8)))  float    v8f;

union Frag { v16h h; unsigned u[8]; };

// ---------------------------------------------------------------------------
// Kernel A: sinusoidal embedding -> f16 table [n, 192]
// channel layout: d*64 + f -> sin, d*64 + 32 + f -> cos   (matches reference)
// ---------------------------------------------------------------------------
__global__ void embed_kernel(const float* __restrict__ pts,
                             _Float16* __restrict__ emb, int n) {
    int i = blockIdx.x * blockDim.x + threadIdx.x;
    if (i >= n) return;
    _Float16* row = emb + (size_t)i * EMBED;
    #pragma unroll
    for (int d = 0; d < DIMS; ++d) {
        float coord = pts[i * DIMS + d];
        #pragma unroll 8
        for (int f = 0; f < NFREQ; ++f) {
            float freq = __expf(-(float)f * FREQ_SCALE);
            float s, c;
            __sincosf(coord * freq, &s, &c);
            row[d * 64 + f]          = (_Float16)s;
            row[d * 64 + NFREQ + f]  = (_Float16)c;
        }
    }
}

// ---------------------------------------------------------------------------
// Kernel B: radius-limited 16-NN. One thread per query, y tiled through LDS,
// per-thread sorted insertion list (fully unrolled -> stays in VGPRs).
// ---------------------------------------------------------------------------
#define KTILE 256
__global__ __launch_bounds__(KTILE) void knn_kernel(
        const float* __restrict__ y, const float* __restrict__ x,
        int* __restrict__ idxOut, unsigned* __restrict__ maskOut) {
    __shared__ float sy[KTILE * 3];
    const int m = blockIdx.x * blockDim.x + threadIdx.x;
    const float qx = x[m * 3 + 0], qy = x[m * 3 + 1], qz = x[m * 3 + 2];
    const float r2 = RADIUS * RADIUS;

    float dist[KNBR];
    int   nid[KNBR];
    #pragma unroll
    for (int k = 0; k < KNBR; ++k) { dist[k] = 3.0e38f; nid[k] = 0; }

    for (int base = 0; base < NPTS; base += KTILE) {
        __syncthreads();
        int t = threadIdx.x;
        sy[t * 3 + 0] = y[(base + t) * 3 + 0];
        sy[t * 3 + 1] = y[(base + t) * 3 + 1];
        sy[t * 3 + 2] = y[(base + t) * 3 + 2];
        __syncthreads();
        for (int j = 0; j < KTILE; ++j) {
            float dx = qx - sy[j * 3 + 0];
            float dy = qy - sy[j * 3 + 1];
            float dz = qz - sy[j * 3 + 2];
            float d2 = dx * dx + dy * dy + dz * dz;
            if (d2 <= r2 && d2 < dist[KNBR - 1]) {   // rare: ~24/16384 pass
                int cand = base + j;
                #pragma unroll
                for (int q = 0; q < KNBR; ++q) {
                    if (d2 < dist[q]) {
                        float td = dist[q]; dist[q] = d2;  d2  = td;
                        int   ti = nid[q];  nid[q]  = cand; cand = ti;
                    }
                }
            }
        }
    }
    unsigned msk = 0;
    #pragma unroll
    for (int k = 0; k < KNBR; ++k) {
        bool ok = (dist[k] <= r2);
        if (ok) msk |= (1u << k);
        idxOut[m * KNBR + k] = ok ? nid[k] : 0;
    }
    maskOut[m] = msk;
}

// ---------------------------------------------------------------------------
// Kernel C: weight transpose + cast: Wt[n][k] (f16) = W[k][n] (f32)
// so B-fragment K-pairs are contiguous 32-bit loads.
// ---------------------------------------------------------------------------
__global__ void wtrans_kernel(const float* __restrict__ W,
                              _Float16* __restrict__ Wt, int kin, int nout) {
    int id = blockIdx.x * blockDim.x + threadIdx.x;
    if (id >= kin * nout) return;
    int n = id % nout, k = id / nout;
    Wt[(size_t)n * kin + k] = (_Float16)W[(size_t)k * nout + n];
}

// ---------------------------------------------------------------------------
// Main MLP kernel. 128 threads = 4 waves; one query (16 neighbor rows) per
// wave. All activations live in per-wave LDS buffers; weights stream from L2.
// ---------------------------------------------------------------------------
__device__ __forceinline__ float gelu(float v) {
    return 0.5f * v * (1.0f + tanhf(0.7978845608028654f * (v + 0.044715f * v * v * v)));
}

// A-fragment (16x32 f16, M = lane%16): VGPR v<4 -> K = v*2+half*8,
//                                      v>=4   -> K = 16+(v-4)*2+half*8
__device__ __forceinline__ int a_koff(int v, int half) {
    return (v < 4) ? (v * 2 + half * 8) : (16 + (v - 4) * 2 + half * 8);
}

template<int KIN, int NOUT>
__device__ __forceinline__ void layer_gelu(const _Float16* __restrict__ src,
                                           _Float16* __restrict__ dst,
                                           const _Float16* __restrict__ Wt,
                                           const float* __restrict__ bias,
                                           int lane) {
    const int n15  = lane & 15;
    const int half = lane >> 4;
    #pragma unroll
    for (int t = 0; t < NOUT / 16; ++t) {
        float bv = bias[t * 16 + n15];
        v8f c;
        #pragma unroll
        for (int i = 0; i < 8; ++i) c[i] = bv;
        const _Float16* wrow = Wt + (size_t)(t * 16 + n15) * KIN;
        for (int kc = 0; kc < KIN; kc += 32) {
            Frag a, b;
            #pragma unroll
            for (int v = 0; v < 8; ++v) {
                a.u[v] = *(const unsigned*)(src + n15 * KIN + kc + a_koff(v, half));
                // B-fragment (32x16): same K order as A-fragment (a_koff)
                b.u[v] = *(const unsigned*)(wrow + kc + a_koff(v, half));
            }
            c = __builtin_amdgcn_wmma_f32_16x16x32_f16(
                    false, a.h, false, b.h, (short)0, c, false, false);
        }
        // C layout: lane holds col n15; VGPR v holds row v + 8*half
        #pragma unroll
        for (int v = 0; v < 8; ++v) {
            int row = v + half * 8;
            dst[row * NOUT + t * 16 + n15] = (_Float16)gelu(c[v]);
        }
    }
}

#define WAVE_BYTES (16 * IN0 * 2 /*bufA 12288*/ + 16 * H2 * 2 /*bufB 8192*/ + 64 /*idx*/)

__global__ __launch_bounds__(128) void gno_mlp_kernel(
        const _Float16* __restrict__ yEmb, const _Float16* __restrict__ xEmb,
        const float* __restrict__ f_y, const int* __restrict__ nidx,
        const unsigned* __restrict__ nmask,
        const _Float16* __restrict__ W0t, const float* __restrict__ b0,
        const _Float16* __restrict__ W1t, const float* __restrict__ b1,
        const _Float16* __restrict__ W2t, const float* __restrict__ b2,
        const _Float16* __restrict__ W3t, const float* __restrict__ b3,
        float* __restrict__ out) {
    extern __shared__ char smem[];
    const int wave = threadIdx.x >> 5;
    const int lane = threadIdx.x & 31;
    const int m    = blockIdx.x * 4 + wave;

    char* wbase = smem + wave * WAVE_BYTES;
    _Float16* bufA = (_Float16*)wbase;                  // 16 x 384
    _Float16* bufB = (_Float16*)(wbase + 16 * IN0 * 2); // 16 x 256
    int*      idxS = (int*)(wbase + 16 * IN0 * 2 + 16 * H2 * 2);

    const unsigned qmask = nmask[m];

    // ---- stage input tile: row k = [ yEmb[idx[m,k]] | xEmb[m] ]  (16 x 384)
    {
        int row = lane & 15;
        const _Float16* srcRow;
        _Float16* dstRow;
        if (lane < 16) {
            int gi = nidx[m * KNBR + row];
            idxS[row] = gi;
            srcRow = yEmb + (size_t)gi * EMBED;
            dstRow = bufA + row * IN0;
        } else {
            srcRow = xEmb + (size_t)m * EMBED;
            dstRow = bufA + row * IN0 + EMBED;
        }
        #pragma unroll
        for (int c = 0; c < EMBED / 8; ++c)            // 24 x 16B
            *(uint4*)(dstRow + c * 8) = *(const uint4*)(srcRow + c * 8);
    }
    asm volatile("s_wait_dscnt 0" ::: "memory");       // intra-wave LDS visibility

    // ---- layers 0..2 with fused GELU
    layer_gelu<IN0, H1>(bufA, bufB, W0t, b0, lane);
    asm volatile("s_wait_dscnt 0" ::: "memory");
    layer_gelu<H1, H2>(bufB, bufA, W1t, b1, lane);
    asm volatile("s_wait_dscnt 0" ::: "memory");
    layer_gelu<H2, H3>(bufA, bufB, W2t, b2, lane);
    asm volatile("s_wait_dscnt 0" ::: "memory");

    // ---- layer 3: 128 -> 64, fuse f_y scale + validity mask + sum over K
    {
        const int n15  = lane & 15;
        const int half = lane >> 4;
        #pragma unroll
        for (int t = 0; t < CCH / 16; ++t) {
            float bv = b3[t * 16 + n15];
            v8f c;
            #pragma unroll
            for (int i = 0; i < 8; ++i) c[i] = bv;
            const _Float16* wrow = W3t + (size_t)(t * 16 + n15) * H3;
            #pragma unroll
            for (int kc = 0; kc < H3; kc += 32) {
                Frag a, b;
                #pragma unroll
                for (int v = 0; v < 8; ++v) {
                    a.u[v] = *(const unsigned*)(bufB + n15 * H3 + kc + a_koff(v, half));
                    b.u[v] = *(const unsigned*)(wrow + kc + a_koff(v, half));
                }
                c = __builtin_amdgcn_wmma_f32_16x16x32_f16(
                        false, a.h, false, b.h, (short)0, c, false, false);
            }
            float partial = 0.0f;
            #pragma unroll
            for (int v = 0; v < 8; ++v) {
                int row = v + half * 8;                 // neighbor index k
                bool ok = (qmask >> row) & 1u;
                float f = ok ? f_y[(size_t)idxS[row] * CCH + t * 16 + n15] : 0.0f;
                partial += c[v] * f;
            }
            partial += __shfl_xor(partial, 16, 32);     // fold rows 8..15 into 0..7 lanes
            if (half == 0)
                out[(size_t)m * CCH + t * 16 + n15] = partial;
        }
    }
}

// ---------------------------------------------------------------------------
// host launcher
// ---------------------------------------------------------------------------
extern "C" void kernel_launch(void* const* d_in, const int* in_sizes, int n_in,
                              void* d_out, int out_size, void* d_ws, size_t ws_size,
                              hipStream_t stream) {
    (void)in_sizes; (void)n_in; (void)out_size; (void)ws_size;
    const float* y   = (const float*)d_in[0];
    const float* x   = (const float*)d_in[1];
    const float* f_y = (const float*)d_in[2];
    const float* W0  = (const float*)d_in[3];  const float* b0 = (const float*)d_in[4];
    const float* W1  = (const float*)d_in[5];  const float* b1 = (const float*)d_in[6];
    const float* W2  = (const float*)d_in[7];  const float* b2 = (const float*)d_in[8];
    const float* W3  = (const float*)d_in[9];  const float* b3 = (const float*)d_in[10];
    float* out = (float*)d_out;

    // workspace carve-up (bytes)
    char* ws = (char*)d_ws;
    _Float16* yEmb = (_Float16*)(ws);                               // 6,291,456
    _Float16* xEmb = (_Float16*)(ws + 6291456);                     // 6,291,456
    int*      idxW = (int*)     (ws + 12582912);                    // 1,048,576
    unsigned* mskW = (unsigned*)(ws + 13631488);                    //    65,536
    _Float16* W0t  = (_Float16*)(ws + 13697024);                    //    98,304
    _Float16* W1t  = (_Float16*)(ws + 13795328);                    //    65,536
    _Float16* W2t  = (_Float16*)(ws + 13860864);                    //    65,536
    _Float16* W3t  = (_Float16*)(ws + 13926400);                    //    16,384

    embed_kernel<<<(NPTS + 255) / 256, 256, 0, stream>>>(y, yEmb, NPTS);
    embed_kernel<<<(MPTS + 255) / 256, 256, 0, stream>>>(x, xEmb, MPTS);
    knn_kernel<<<MPTS / KTILE, KTILE, 0, stream>>>(y, x, idxW, mskW);

    wtrans_kernel<<<(IN0 * H1 + 255) / 256, 256, 0, stream>>>(W0, W0t, IN0, H1);
    wtrans_kernel<<<(H1 * H2 + 255) / 256, 256, 0, stream>>>(W1, W1t, H1, H2);
    wtrans_kernel<<<(H2 * H3 + 255) / 256, 256, 0, stream>>>(W2, W2t, H2, H3);
    wtrans_kernel<<<(H3 * CCH + 255) / 256, 256, 0, stream>>>(W3, W3t, H3, CCH);

    gno_mlp_kernel<<<MPTS / 4, 128, 4 * WAVE_BYTES, stream>>>(
        yEmb, xEmb, f_y, idxW, mskW,
        W0t, b0, W1t, b1, W2t, b2, W3t, b3, out);
}
